// Encoder_20658792694195
// MI455X (gfx1250) — matrix operand order repaired and checked by reading, not compiled, not run on hardware
//
#include <hip/hip_runtime.h>

// ---------------------------------------------------------------------------
// Types for CDNA5 WMMA (wave32, 16x16x32 bf16 -> f32)
// ---------------------------------------------------------------------------
typedef __attribute__((ext_vector_type(16))) __bf16 v16bf;
typedef __attribute__((ext_vector_type(8)))  __bf16 v8bf;
typedef __attribute__((ext_vector_type(8)))  float  v8f;

// Problem constants
#define BZ 64
#define TT 256
#define EE 512
#define UU 1024
#define GG 4096   // 4*U

// LDS A-panel row padding (elements) to break the 256B bank period
#define APAD 8

// 32-bit LDS byte offset of a pointer into a __shared__ object
// (addrspace(3) pointers on amdgcn are 32-bit offsets from LDS base)
__device__ __forceinline__ unsigned lds_off_u32(const void* p) {
    return (unsigned)(unsigned long long)
        (__attribute__((address_space(3))) const void*)p;
}

// ---------------------------------------------------------------------------
// Cast f32 weight [K,N] (row-major) -> bf16 transposed [N,K] (row-major)
// ---------------------------------------------------------------------------
__global__ void cast_transpose(const float* __restrict__ W, __bf16* __restrict__ Wt,
                               int K, int N) {
    __shared__ float tile[32][33];
    const int kb = blockIdx.x * 32;
    const int nb = blockIdx.y * 32;
    const int tx = threadIdx.x & 31;
    const int ty = threadIdx.x >> 5;        // 0..7
    #pragma unroll
    for (int i = ty; i < 32; i += 8)
        tile[i][tx] = W[(size_t)(kb + i) * N + (nb + tx)];
    __syncthreads();
    #pragma unroll
    for (int i = ty; i < 32; i += 8)
        Wt[(size_t)(nb + i) * K + (kb + tx)] = (__bf16)tile[tx][i];
}

// ---------------------------------------------------------------------------
// Embedding gather + cast to bf16.  One block per (b,t) row.
// ---------------------------------------------------------------------------
__global__ void embed_cast(const int* __restrict__ tokens, const float* __restrict__ emb,
                           __bf16* __restrict__ x0) {
    const int row = blockIdx.x;                 // b*T + t
    const int tok = tokens[row];
    const float* src = emb + (size_t)tok * EE;
    __bf16* dst = x0 + (size_t)row * EE;
    for (int e = threadIdx.x; e < EE; e += blockDim.x)
        dst[e] = (__bf16)src[e];
}

// ---------------------------------------------------------------------------
// Zero recurrent state + sync counter
// ---------------------------------------------------------------------------
__global__ void init_state(__bf16* __restrict__ hbuf, float* __restrict__ cbuf,
                           unsigned* __restrict__ counter) {
    const int i = blockIdx.x * blockDim.x + threadIdx.x;
    if (i < 2 * BZ * UU) hbuf[i] = (__bf16)0.0f;
    if (i < BZ * UU)     cbuf[i] = 0.0f;
    if (i == 0)          *counter = 0u;
}

// ---------------------------------------------------------------------------
// GEMM: C[M,N] f32 = A[M,K]bf16(row-major) x Bt[N,K]bf16 (B column-major) + bias
// Block = 8 waves = 16x128 C tile (all waves share one A K-panel).
// A K-panel (16 x K bf16, contiguous rows) staged once into LDS via
// global_load_async_to_lds_b128 (ASYNCcnt), then consumed as ds_load_b128
// WMMA fragments.  B streams from L2.
// Fragment layouts loaded here (contiguous k, not the ISA table order):
//   A 16x32 bf16: half=lane>>4, row=lane&15, contiguous K-chunk kb+16h..+15 (from LDS row)
//   B 32x16 bf16: col=lane&15, contiguous K-chunk kb+16h..+15  (from Bt row)
//   C/D f32:      row = (lane>>4)*8 + i, col = lane&15
// ---------------------------------------------------------------------------
__global__ void gemm_bf16_wmma(const __bf16* __restrict__ A, const __bf16* __restrict__ Bt,
                               const float* __restrict__ bias, float* __restrict__ C,
                               int M, int N, int K) {
    __shared__ __align__(16) __bf16 apanel[16 * (1024 + APAD)];
    const int lane = threadIdx.x & 31;
    const int wave = threadIdx.x >> 5;
    const int tm   = blockIdx.x;                  // M/16
    const int tn   = blockIdx.y * 8 + wave;       // N/16
    const int half = lane >> 4;
    const int l15  = lane & 15;
    const int krow = K + APAD;                    // padded LDS row stride (elems)

    // ---- async stage of the 16 x K A-panel into LDS (rows are contiguous) ----
    {
        const __bf16* src = A + (size_t)tm * 16 * K;
        const int cpr     = K >> 3;               // 16B chunks per row
        const int total   = 16 * cpr;
        for (int c = threadIdx.x; c < total; c += 256) {
            const int row = c / cpr;
            const int j   = c - row * cpr;
            const __bf16* g = src + (size_t)row * K + j * 8;
            const unsigned lds_addr =
                lds_off_u32(&apanel[row * krow + j * 8]);
            asm volatile("global_load_async_to_lds_b128 %0, %1, off"
                         :: "v"(lds_addr), "v"(g) : "memory");
        }
        asm volatile("s_wait_asynccnt 0" ::: "memory");
    }
    __syncthreads();

    const __bf16* Alds = apanel + l15 * krow + half * 16;
    const __bf16* Bp   = Bt + (size_t)(tn * 16 + l15) * K + half * 16;

    v8f acc = {};
    #pragma unroll 4
    for (int k = 0; k < K; k += 32) {
        v8bf a0 = *(const v8bf*)(Alds + k);       // ds_load_b128
        v8bf a1 = *(const v8bf*)(Alds + k + 8);   // ds_load_b128
        v16bf a = __builtin_shufflevector(a0, a1, 0,1,2,3,4,5,6,7,8,9,10,11,12,13,14,15);
        v16bf b = *(const v16bf*)(Bp + k);        // 2x global_load_b128
        acc = __builtin_amdgcn_wmma_f32_16x16x32_bf16(false, a, false, b,
                                                      (short)0, acc, false, false);
    }
    const int ncol = tn * 16 + l15;
    const float bv = bias ? bias[ncol] : 0.0f;
    float* Cp = C + (size_t)(tm * 16 + half * 8) * N + ncol;
    #pragma unroll
    for (int i = 0; i < 8; ++i)
        Cp[(size_t)i * N] = acc[i] + bv;
}

// ---------------------------------------------------------------------------
// Persistent LSTM layer.  Grid = 64 blocks x 256 threads (co-resident).
// Block g owns units u0 = g*16 .. u0+15 (all 4 gate columns, full batch).
// Per step: 16 WMMA tiles (4 batch-tiles x 4 gates), 2 per wave, K=1024.
// h double-buffered bf16 in global; gate fusion through LDS; grid barrier
// via monotonic atomic counter + s_sleep spin.
// ---------------------------------------------------------------------------
__global__ void lstm_layer(const float* __restrict__ xz,      // [(b*T+t), 4096] f32
                           const __bf16* __restrict__ Ut,     // [4096][1024] bf16
                           __bf16* __restrict__ hbuf,         // [2][64][1024] bf16
                           float* __restrict__ cbuf,          // [64][1024] f32
                           __bf16* __restrict__ seq_bf,       // layer0 out (or null)
                           float* __restrict__ seq_f32,       // layer1 out (or null)
                           float* __restrict__ hout,          // [64][1024] or null
                           float* __restrict__ cout,          // [64][1024] or null
                           unsigned* __restrict__ counter) {
    __shared__ float zlds[4][4][16][16];       // [btile][gate][m][n] = 16 KB
    const int lane = threadIdx.x & 31;
    const int wave = threadIdx.x >> 5;
    const int half = lane >> 4;
    const int l15  = lane & 15;
    const int u0   = blockIdx.x * 16;

    for (int t = 0; t < TT; ++t) {
        const __bf16* hprev = hbuf + (size_t)(t & 1) * (BZ * UU);
        __bf16* hnext       = hbuf + (size_t)((t + 1) & 1) * (BZ * UU);

        #pragma unroll
        for (int rep = 0; rep < 2; ++rep) {
            const int tid   = wave + rep * 8;  // 0..15
            const int btile = tid >> 2;
            const int gate  = tid & 3;
            const int col   = gate * UU + u0 + l15;     // z column == Ut row
            const __bf16* Ap = hprev + (size_t)(btile * 16 + l15) * UU + half * 16;
            const __bf16* Bp = Ut + (size_t)col * UU + half * 16;

            v8f acc = {};
            #pragma unroll 4
            for (int k = 0; k < UU; k += 32) {
                v8bf a0 = *(const v8bf*)(Ap + k);
                v8bf a1 = *(const v8bf*)(Ap + k + 8);
                v16bf a = __builtin_shufflevector(a0, a1,
                            0,1,2,3,4,5,6,7,8,9,10,11,12,13,14,15);
                v16bf b = *(const v16bf*)(Bp + k);
                acc = __builtin_amdgcn_wmma_f32_16x16x32_bf16(false, a, false, b,
                                                              (short)0, acc, false, false);
            }
            // add precomputed input projection, stash pre-activation tile in LDS
            #pragma unroll
            for (int i = 0; i < 8; ++i) {
                const int m = half * 8 + i;
                const int b = btile * 16 + m;
                const float z = acc[i] +
                    xz[((size_t)b * TT + t) * GG + gate * UU + u0 + l15];
                zlds[btile][gate][m][l15] = z;
            }
        }
        __syncthreads();

        // Elementwise gate fusion: 64 batch x 16 units = 1024 cells, 4/thread
        for (int r = threadIdx.x; r < BZ * 16; r += 256) {
            const int b  = r >> 4;
            const int un = r & 15;
            const int bt = b >> 4, m = b & 15;
            float iv = zlds[bt][0][m][un];
            float fv = zlds[bt][1][m][un];
            float gv = zlds[bt][2][m][un];
            float ov = zlds[bt][3][m][un];
            iv = 1.0f / (1.0f + __expf(-iv));
            fv = 1.0f / (1.0f + __expf(-fv));
            gv = tanhf(gv);
            ov = 1.0f / (1.0f + __expf(-ov));
            const int u = u0 + un;
            const size_t bu = (size_t)b * UU + u;
            const float c = fv * cbuf[bu] + iv * gv;
            const float h = ov * tanhf(c);
            cbuf[bu]  = c;
            hnext[bu] = (__bf16)h;
            if (seq_bf)  seq_bf[((size_t)b * TT + t) * UU + u] = (__bf16)h;
            if (seq_f32) seq_f32[((size_t)b * TT + t) * UU + u] = h;
            if (hout && t == TT - 1) { hout[bu] = h; cout[bu] = c; }
        }
        __syncthreads();

        // Grid-wide step barrier (monotonic counter, no reset)
        if (threadIdx.x == 0) {
            __threadfence();
            atomicAdd(counter, 1u);
            const unsigned target = 64u * (unsigned)(t + 1);
            while (atomicAdd(counter, 0u) < target)
                __builtin_amdgcn_s_sleep(1);
        }
        __syncthreads();
        __threadfence();   // acquire: invalidate near caches before reading new h
    }
}

// ---------------------------------------------------------------------------
// Host launcher
// ---------------------------------------------------------------------------
extern "C" void kernel_launch(void* const* d_in, const int* in_sizes, int n_in,
                              void* d_out, int out_size, void* d_ws, size_t ws_size,
                              hipStream_t stream) {
    const int*   tokens = (const int*)d_in[0];
    const float* emb    = (const float*)d_in[1];
    const float* W0     = (const float*)d_in[2];
    const float* U0     = (const float*)d_in[3];
    const float* b0     = (const float*)d_in[4];
    const float* W1     = (const float*)d_in[5];
    const float* U1     = (const float*)d_in[6];
    const float* b1     = (const float*)d_in[7];

    const size_t BT = (size_t)BZ * TT;

    char* p = (char*)d_ws;
    auto alloc = [&](size_t bytes) -> void* {
        void* r = (void*)p;
        p += (bytes + 255) & ~(size_t)255;
        return r;
    };
    __bf16* W0t  = (__bf16*)alloc((size_t)GG * EE * 2);
    __bf16* U0t  = (__bf16*)alloc((size_t)GG * UU * 2);
    __bf16* W1t  = (__bf16*)alloc((size_t)GG * UU * 2);
    __bf16* U1t  = (__bf16*)alloc((size_t)GG * UU * 2);
    __bf16* x0   = (__bf16*)alloc(BT * EE * 2);
    __bf16* x1   = (__bf16*)alloc(BT * UU * 2);
    float*  xz   = (float*)alloc(BT * GG * 4);          // shared by both layers
    __bf16* hbuf = (__bf16*)alloc((size_t)2 * BZ * UU * 2);
    float*  cbuf = (float*)alloc((size_t)BZ * UU * 4);
    unsigned* counter = (unsigned*)alloc(256);

    const dim3 t256(256);

    // Weight casts/transposes (one-time per launch, ~28 MB)
    cast_transpose<<<dim3(EE / 32, GG / 32), t256, 0, stream>>>(W0, W0t, EE, GG);
    cast_transpose<<<dim3(UU / 32, GG / 32), t256, 0, stream>>>(U0, U0t, UU, GG);
    cast_transpose<<<dim3(UU / 32, GG / 32), t256, 0, stream>>>(W1, W1t, UU, GG);
    cast_transpose<<<dim3(UU / 32, GG / 32), t256, 0, stream>>>(U1, U1t, UU, GG);

    // Embedding gather
    embed_cast<<<dim3((unsigned)BT), t256, 0, stream>>>(tokens, emb, x0);

    // ---- Layer 0 ----
    gemm_bf16_wmma<<<dim3((unsigned)(BT / 16), GG / 128), t256, 0, stream>>>(
        x0, W0t, b0, xz, (int)BT, GG, EE);
    init_state<<<512, t256, 0, stream>>>(hbuf, cbuf, counter);
    lstm_layer<<<64, t256, 0, stream>>>(xz, U0t, hbuf, cbuf,
                                        x1, nullptr, nullptr, nullptr, counter);

    // ---- Layer 1 ----
    gemm_bf16_wmma<<<dim3((unsigned)(BT / 16), GG / 128), t256, 0, stream>>>(
        x1, W1t, b1, xz, (int)BT, GG, UU);
    init_state<<<512, t256, 0, stream>>>(hbuf, cbuf, counter);
    float* seq  = (float*)d_out;
    float* hout = seq + BT * UU;
    float* cout = hout + (size_t)BZ * UU;
    lstm_layer<<<64, t256, 0, stream>>>(xz, U1t, hbuf, cbuf,
                                        nullptr, seq, hout, cout, counter);
}
